// WindowAttention_15247133900942
// MI455X (gfx1250) — matrix operand order repaired and checked by reading, not compiled, not run on hardware
//
#include <hip/hip_runtime.h>

typedef __bf16 bf16;
typedef __attribute__((ext_vector_type(16))) __bf16 bf16x16;
typedef __attribute__((ext_vector_type(8)))  __bf16 bf16x8;
typedef __attribute__((ext_vector_type(2)))  __bf16 bf16x2;
typedef __attribute__((ext_vector_type(8)))  float  f32x8;
typedef __attribute__((ext_vector_type(4)))  float  f32x4;

namespace {
constexpr int NTOK  = 49;    // window tokens (7x7)
constexpr int NPAD  = 64;    // padded to 4 WMMA tiles
constexpr int HEADS = 8;
constexpr int DIMC  = 256;
constexpr int HDIM  = 32;
constexpr int NWIN  = 64;    // mask windows
constexpr int BWIN  = 4096;  // batch of windows
constexpr float SCALE = 0.17677669529663687f; // 32^-0.5
}

// ---- WMMA fragment builders (wave32; layouts per CDNA5 ISA 7.12.2) ----

// A fragment 16x32 bf16 built from fp32 row-major source (converts on the fly).
// Per-lane: row = lane&15 (baked into rowp); two contiguous runs of 8:
//   K = kbase + half*8 + [0..7]  and  K = kbase + 16 + half*8 + [0..7]
__device__ __forceinline__ bf16x16 afrag_f32(const float* rowp, int kbase) {
  const int half = ((int)threadIdx.x >> 4) & 1;
  const float* p = rowp + kbase + half * 8;
  f32x4 a0 = *(const f32x4*)(p);
  f32x4 a1 = *(const f32x4*)(p + 4);
  f32x4 a2 = *(const f32x4*)(p + 16);
  f32x4 a3 = *(const f32x4*)(p + 20);
  bf16x16 r;
#pragma unroll
  for (int i = 0; i < 4; ++i) {
    r[i]      = (bf16)a0[i];
    r[4 + i]  = (bf16)a1[i];
    r[8 + i]  = (bf16)a2[i];
    r[12 + i] = (bf16)a3[i];
  }
  return r;
}

// A fragment 16x32 bf16 from bf16 row-major source (LDS or global).
__device__ __forceinline__ bf16x16 afrag_bf16(const bf16* rowp, int kbase) {
  const int half = ((int)threadIdx.x >> 4) & 1;
  const bf16* p = rowp + kbase + half * 8;
  bf16x8 lo = *(const bf16x8*)(p);
  bf16x8 hi = *(const bf16x8*)(p + 16);
  bf16x16 r;
#pragma unroll
  for (int i = 0; i < 8; ++i) { r[i] = lo[i]; r[8 + i] = hi[i]; }
  return r;
}

// B fragment 32x16 bf16 from a [col][K] (transposed-B) bf16 source.
// Per-lane: col = lane&15 (baked into colp); two runs of 8: K = kbase + half*8 + [0..7] and kbase + 16 + half*8 + [0..7].
__device__ __forceinline__ bf16x16 bfrag_bf16(const bf16* colp, int kbase) {
  const int half = ((int)threadIdx.x >> 4) & 1;
  const bf16* p = colp + kbase + half * 8;
  bf16x8 lo = *(const bf16x8*)(p);
  bf16x8 hi = *(const bf16x8*)(p + 16);
  bf16x16 r;
#pragma unroll
  for (int i = 0; i < 8; ++i) { r[i] = lo[i]; r[8 + i] = hi[i]; }
  return r;
}

// ---- kernel 0: weight fp32 -> bf16 conversion ----
__global__ void cvt_weights_kernel(const float* __restrict__ qkv_w,
                                   const float* __restrict__ proj_w,
                                   bf16* __restrict__ wqkv,
                                   bf16* __restrict__ wproj) {
  const int i = blockIdx.x * blockDim.x + threadIdx.x;
  if (i < 3 * DIMC * DIMC) wqkv[i] = (bf16)qkv_w[i];
  if (i < DIMC * DIMC)     wproj[i] = (bf16)proj_w[i];
}

// ---- kernel 1: fused QKV + attention per window ----
__global__ __launch_bounds__(256) void win_attn_kernel(
    const float* __restrict__ x, const float* __restrict__ mask,
    const bf16* __restrict__ wqkv, const float* __restrict__ qkv_b,
    const float* __restrict__ bias_table,
    float* __restrict__ attn_out, bf16* __restrict__ obuf) {
  __shared__ bf16 sQ[NPAD * 40];    // q * SCALE, [n][d], stride 40
  __shared__ bf16 sK[NPAD * 40];    // k, [n][d], stride 40
  __shared__ bf16 sVT[HDIM * 72];   // v transposed [d][n], stride 72
  __shared__ float sS[NPAD * 65];   // scores fp32, stride 65 (bank-friendly)
  __shared__ bf16 sP[NPAD * 72];    // softmax probs bf16, stride 72
  __shared__ float sMax[NPAD];
  __shared__ float sInv[NPAD];

  const int b = blockIdx.x;
  const int w = b & (NWIN - 1);     // mask window index = b % 64
  const int tid = (int)threadIdx.x;
  // readfirstlane: prove wave-uniformity to the compiler -> scalar control
  // flow (s_cbranch) around WMMA, EXEC stays all-ones as the ISA requires.
  const int wave = __builtin_amdgcn_readfirstlane(tid >> 5);
  const int lane = tid & 31;
  const int col16 = lane & 15;
  const int half = lane >> 4;

  const float* xw = x + (size_t)b * NTOK * DIMC;
  const float* maskw = mask + (size_t)w * NTOK * NTOK;

  // prefetch the x window (emits global_prefetch_b8); 128B per lane
  for (int off = tid * 32; off < NTOK * DIMC; off += 256 * 32)
    __builtin_prefetch(xw + off, 0, 1);

  for (int h = 0; h < HEADS; ++h) {
    // ---- QKV GEMM for head h: M=64(pad), N=96 (q|k|v, 32 each), K=256 ----
    // 24 output tiles; wave handles t = wave, wave+8, wave+16 (all scalar).
#pragma unroll
    for (int i = 0; i < 3; ++i) {
      const int t = wave + 8 * i;
      const int tm = t & 3;          // M tile (scalar)
      const int tc = t >> 2;         // 0..5   (scalar)
      const int which = tc >> 1;     // 0=q, 1=k, 2=v (scalar)
      const int sub = tc & 1;        // 16-col half of head dim (scalar)
      const int col = which * DIMC + h * HDIM + sub * 16 + col16;
      const float bv = qkv_b[col];
      f32x8 acc = {bv, bv, bv, bv, bv, bv, bv, bv};
      const int arow = tm * 16 + col16;
      const int arc = arow < NTOK ? arow : NTOK - 1; // clamp pad rows
      const float* rowp = xw + (size_t)arc * DIMC;
      const bf16* colp = wqkv + (size_t)col * DIMC;
#pragma unroll
      for (int ks = 0; ks < 8; ++ks) {
        bf16x16 a  = afrag_f32(rowp, ks * 32);
        bf16x16 bb = bfrag_bf16(colp, ks * 32);
        acc = __builtin_amdgcn_wmma_f32_16x16x32_bf16(
            false, a, false, bb, (short)0, acc, false, false);
      }
      const int dbase = sub * 16 + col16;
      if (which == 0) {              // scalar branch
#pragma unroll
        for (int r = 0; r < 8; ++r)
          sQ[(tm * 16 + r + half * 8) * 40 + dbase] = (bf16)(acc[r] * SCALE);
      } else if (which == 1) {
#pragma unroll
        for (int r = 0; r < 8; ++r)
          sK[(tm * 16 + r + half * 8) * 40 + dbase] = (bf16)acc[r];
      } else {                       // v: transposed, packed pair stores
#pragma unroll
        for (int r = 0; r < 8; r += 2) {
          bf16x2 pk = {(bf16)acc[r], (bf16)acc[r + 1]};
          *(bf16x2*)&sVT[dbase * 72 + tm * 16 + r + half * 8] = pk;
        }
      }
    }
    __syncthreads();

    // ---- S = (q*SCALE) @ k^T : 16 tiles (64x64 pad), 2 per wave, K=32 ----
#pragma unroll
    for (int i = 0; i < 2; ++i) {
      const int t = wave * 2 + i;    // scalar
      const int tm = t >> 2;
      const int tn = t & 3;
      f32x8 acc = {0.f, 0.f, 0.f, 0.f, 0.f, 0.f, 0.f, 0.f};
      bf16x16 a  = afrag_bf16(sQ + (tm * 16 + col16) * 40, 0);
      bf16x16 bb = bfrag_bf16(sK + (tn * 16 + col16) * 40, 0);
      acc = __builtin_amdgcn_wmma_f32_16x16x32_bf16(
          false, a, false, bb, (short)0, acc, false, false);
#pragma unroll
      for (int r = 0; r < 8; ++r) {
        const int n = tm * 16 + r + half * 8;
        const int m = tn * 16 + col16;
        float sv = -1e30f;
        if (n < NTOK && m < NTOK) {
          // relative position bias index, computed arithmetically (== REL_IDX)
          const int nh = n / 7, nw = n - nh * 7;
          const int mh = m / 7, mw = m - mh * 7;
          const int idx = (nh - mh + 6) * 13 + (nw - mw + 6);
          sv = acc[r] + bias_table[idx * HEADS + h] + maskw[n * NTOK + m];
        }
        sS[n * 65 + m] = sv;
      }
    }
    __syncthreads();

    // ---- softmax stage 1: per-row max & 1/sum (64 threads) ----
    if (tid < NPAD) {
      const int n = tid;
      float mx = -1e30f;
      for (int m = 0; m < NTOK; ++m) mx = fmaxf(mx, sS[n * 65 + m]);
      float sum = 0.f;
      for (int m = 0; m < NTOK; ++m) sum += __expf(sS[n * 65 + m] - mx);
      sMax[n] = mx;
      sInv[n] = 1.0f / sum;
    }
    __syncthreads();

    // ---- softmax stage 2: all 256 threads write P (LDS) + attn (global) ----
    {
      float* ablk = attn_out + ((size_t)b * HEADS + h) * NTOK * NTOK;
#pragma unroll
      for (int i = 0; i < 16; ++i) {       // 64*64 / 256
        const int e = tid + i * 256;
        const int n = e >> 6;
        const int m = e & 63;
        float p = 0.f;
        if (n < NTOK && m < NTOK) {
          p = __expf(sS[n * 65 + m] - sMax[n]) * sInv[n];
          ablk[n * NTOK + m] = p;
        }
        sP[n * 72 + m] = (bf16)p;
      }
    }
    __syncthreads();

    // ---- O_h = P @ V : 8 tiles (64x32), 1 per wave, K=64 (2 steps) ----
    {
      const int tm = wave >> 1;      // scalar
      const int tn = wave & 1;       // scalar
      f32x8 acc = {0.f, 0.f, 0.f, 0.f, 0.f, 0.f, 0.f, 0.f};
#pragma unroll
      for (int ks = 0; ks < 2; ++ks) {
        bf16x16 a  = afrag_bf16(sP + (tm * 16 + col16) * 72, ks * 32);
        bf16x16 bb = bfrag_bf16(sVT + (tn * 16 + col16) * 72, ks * 32);
        acc = __builtin_amdgcn_wmma_f32_16x16x32_bf16(
            false, a, false, bb, (short)0, acc, false, false);
      }
#pragma unroll
      for (int r = 0; r < 8; ++r) {
        const int n = tm * 16 + r + half * 8;
        if (n < NTOK)
          obuf[((size_t)b * NTOK + n) * DIMC + h * HDIM + tn * 16 + col16] =
              (bf16)acc[r];
      }
    }
    __syncthreads();
  }
}

// ---- kernel 2: out = o @ proj_w^T + proj_b ; M=200704=64*3136, N=K=256 ----
__global__ __launch_bounds__(256) void proj_kernel(
    const bf16* __restrict__ obuf, const bf16* __restrict__ wproj,
    const float* __restrict__ proj_b, float* __restrict__ out) {
  const int tid = (int)threadIdx.x;
  const int wave = __builtin_amdgcn_readfirstlane(tid >> 5);
  const int lane = tid & 31;
  const int col16 = lane & 15;
  const int half = lane >> 4;
  const size_t rowbase = (size_t)blockIdx.x * 64;

#pragma unroll
  for (int i = 0; i < 8; ++i) {          // 64 tiles, 8 per wave (scalar t)
    const int t = wave + 8 * i;
    const int tm = t & 3;
    const int tn = t >> 2;
    const int col = tn * 16 + col16;
    const float bv = proj_b[col];
    f32x8 acc = {bv, bv, bv, bv, bv, bv, bv, bv};
    const bf16* rowp = obuf + (rowbase + (size_t)(tm * 16 + col16)) * DIMC;
    const bf16* colp = wproj + (size_t)col * DIMC;
#pragma unroll
    for (int ks = 0; ks < 8; ++ks) {
      bf16x16 a  = afrag_bf16(rowp, ks * 32);
      bf16x16 bb = bfrag_bf16(colp, ks * 32);
      acc = __builtin_amdgcn_wmma_f32_16x16x32_bf16(
          false, a, false, bb, (short)0, acc, false, false);
    }
#pragma unroll
    for (int r = 0; r < 8; ++r) {
      out[(rowbase + (size_t)(tm * 16 + r + half * 8)) * DIMC + col] = acc[r];
    }
  }
}

extern "C" void kernel_launch(void* const* d_in, const int* in_sizes, int n_in,
                              void* d_out, int out_size, void* d_ws, size_t ws_size,
                              hipStream_t stream) {
  const float* x          = (const float*)d_in[0];
  const float* mask       = (const float*)d_in[1];
  const float* qkv_w      = (const float*)d_in[2];
  const float* qkv_b      = (const float*)d_in[3];
  const float* proj_w     = (const float*)d_in[4];
  const float* proj_b     = (const float*)d_in[5];
  const float* bias_table = (const float*)d_in[6];

  // workspace layout: [wqkv bf16 | wproj bf16 | o bf16]
  char* ws = (char*)d_ws;
  bf16* wqkv  = (bf16*)ws;                                    // 768*256
  bf16* wproj = (bf16*)(ws + (size_t)3 * DIMC * DIMC * 2);    // 256*256
  bf16* obuf  = (bf16*)(ws + (size_t)4 * DIMC * DIMC * 2);    // 200704*256

  float* out = (float*)d_out;
  float* attn_out = out + (size_t)BWIN * NTOK * DIMC;         // tuple elem 1

  cvt_weights_kernel<<<(3 * DIMC * DIMC + 255) / 256, 256, 0, stream>>>(
      qkv_w, proj_w, wqkv, wproj);
  win_attn_kernel<<<BWIN, 256, 0, stream>>>(
      x, mask, wqkv, qkv_b, bias_table, attn_out, obuf);
  proj_kernel<<<BWIN * NTOK / 64, 256, 0, stream>>>(obuf, wproj, proj_b, out);
}
